// GCN_28930899706010
// MI455X (gfx1250) — compile-verified, matrix operand order checked by reading, not run on hardware
//
#include <hip/hip_runtime.h>
#include <hip/hip_bf16.h>

typedef __attribute__((ext_vector_type(16))) _Float16 v16h;
typedef __attribute__((ext_vector_type(8)))  float    v8f;

#define N_NODES   50000
#define N_EDGES   800000
#define IN_FEATS  128
#define SEQ_LEN   20
#define N_CLASSES 40

// ---------------------------------------------------------------- utilities
__global__ void zero_f32_kernel(float* __restrict__ p, int n) {
    int i = blockIdx.x * blockDim.x + threadIdx.x;
    if (i < n) p[i] = 0.0f;
}

// out_deg[src[e]] += 1 ; in_deg[dst[e]] += 1
__global__ void degree_kernel(const int* __restrict__ src, const int* __restrict__ dst,
                              float* __restrict__ outd, float* __restrict__ ind, int n_edges) {
    int e = blockIdx.x * blockDim.x + threadIdx.x;
    if (e < n_edges) {
        atomicAdd(&outd[src[e]], 1.0f);
        atomicAdd(&ind[dst[e]], 1.0f);
    }
}

// deg -> rsqrt(max(deg,1)) in place, for both arrays
__global__ void finalize_deg_kernel(float* __restrict__ outd, float* __restrict__ ind, int n) {
    int i = blockIdx.x * blockDim.x + threadIdx.x;
    if (i < n) {
        outd[i] = rsqrtf(fmaxf(outd[i], 1.0f));
        ind[i]  = rsqrtf(fmaxf(ind[i], 1.0f));
    }
}

// Pack f32 weight [K=128 x Nin] row-major into WMMA B-fragment-major f16:
// out[((ntile*4 + kb)*32 + lane)*16 + e], where element e of lane's fragment is
// B[k][n] with k = kb*32 + e%8 + 8*(lane/16) + 16*(e/8), n = ntile*16 + lane%16.
// Zero-pads columns n >= Nin. One thread per packed element.
__global__ void pack_w_kernel(const float* __restrict__ in, _Float16* __restrict__ out,
                              int Nin, int Ntiles) {
    int idx = blockIdx.x * blockDim.x + threadIdx.x;
    int total = Ntiles * 4 * 32 * 16;
    if (idx >= total) return;
    int e     = idx & 15;
    int lane  = (idx >> 4) & 31;
    int kb    = (idx >> 9) & 3;
    int ntile = idx >> 11;
    int k = kb * 32 + (e & 7) + 8 * (lane >> 4) + 16 * (e >> 3);
    int n = ntile * 16 + (lane & 15);
    out[idx] = (n < Nin) ? (_Float16)in[k * Nin + n] : (_Float16)0.0f;
}

// ---------------------------------------------------------------- embedding mean-pool
// one block (128 threads) per node; each thread owns one feature dim
__global__ void embed_pool_kernel(const int* __restrict__ feats,
                                  const float* __restrict__ emb,
                                  float* __restrict__ h) {
    int node = blockIdx.x;
    int f = threadIdx.x;
    float sum = 0.0f;
    int cnt = 0;
    #pragma unroll
    for (int t = 0; t < SEQ_LEN; ++t) {
        int tok = feats[node * SEQ_LEN + t];   // broadcast across the block
        if (tok != 0) {                        // pad row (ix 0) is zero anyway
            ++cnt;
            sum += emb[(size_t)tok * IN_FEATS + f];
        }
    }
    float inv = 1.0f / (float)(cnt > 0 ? cnt : 1);
    h[(size_t)node * IN_FEATS + f] = sum * inv;
}

// ---------------------------------------------------------------- edge scatter (SpMM)
// one wave per edge; each lane: float4 gather, scale by rsqrt(out_deg[src]), 4 f32 atomics
__global__ void scatter_kernel(const int* __restrict__ src, const int* __restrict__ dst,
                               const float* __restrict__ h, const float* __restrict__ so,
                               float* __restrict__ agg, int n_edges) {
    int lane = threadIdx.x & 31;
    int e = blockIdx.x * (blockDim.x >> 5) + (threadIdx.x >> 5);
    if (e >= n_edges) return;
    int s = src[e], d = dst[e];
    float sc = so[s];
    const float4* hv = reinterpret_cast<const float4*>(h + (size_t)s * IN_FEATS);
    float4 v = hv[lane];
    float* out = agg + (size_t)d * IN_FEATS + lane * 4;
    atomicAdd(out + 0, v.x * sc);
    atomicAdd(out + 1, v.y * sc);
    atomicAdd(out + 2, v.z * sc);
    atomicAdd(out + 3, v.w * sc);
}

// ---------------------------------------------------------------- WMMA GEMM
// out[m,n] = act( sum_k (in[m,k]*si[m]) * W[k,n] + bias[n] )
// K = 128. One wave per 16x16 tile: blockDim = 32*Ntiles, gridDim = M/16.
// B fragments come pre-packed (one contiguous 32B load per lane per k-block).
__global__ void gemm_wmma_kernel(const float* __restrict__ in, const float* __restrict__ si,
                                 const v16h* __restrict__ Wpack, const float* __restrict__ bias,
                                 float* __restrict__ out,
                                 int Nout, int out_stride, int relu) {
    const int lane  = threadIdx.x & 31;
    const int ntile = threadIdx.x >> 5;
    const int mtile = blockIdx.x;
    const int l16   = lane >> 4;   // 0: k 0-7,16-23; 1: k 8-15,24-31 of each k-block
    const int lm    = lane & 15;

    const int arow = mtile * 16 + lm;          // A-matrix row owned by this lane
    const float sA = si[arow];
    const float* arow_p = in + (size_t)arow * IN_FEATS;

    v8f c = {};
    #pragma unroll
    for (int kb = 0; kb < 4; ++kb) {           // 4 x (16x16x32) over K=128
        // A fragment: two contiguous 8-float chunks per lane (vectorizes to b128 pairs)
        const float* ap = arow_p + kb * 32 + 8 * l16;
        v16h a;
        #pragma unroll
        for (int e = 0; e < 8; ++e) a[e]     = (_Float16)(ap[e] * sA);
        #pragma unroll
        for (int e = 0; e < 8; ++e) a[8 + e] = (_Float16)(ap[16 + e] * sA);
        // B fragment: one contiguous 32-byte load
        v16h b = Wpack[(ntile * 4 + kb) * 32 + lane];
        c = __builtin_amdgcn_wmma_f32_16x16x32_f16(
                /*neg_a=*/false, a, /*neg_b=*/false, b,
                /*c_mod=*/(short)0, c, /*reuse_a=*/false, /*reuse_b=*/false);
    }

    // C/D layout: vgpr v, lane l -> M = v + 8*(l/16), N = l%16
    #pragma unroll
    for (int v = 0; v < 8; ++v) {
        int m = mtile * 16 + v + 8 * l16;
        int n = ntile * 16 + lm;
        if (n < Nout) {
            float val = c[v] + bias[n];
            if (relu) val = fmaxf(val, 0.0f);
            out[(size_t)m * out_stride + n] = val;
        }
    }
}

// ---------------------------------------------------------------- launch
extern "C" void kernel_launch(void* const* d_in, const int* in_sizes, int n_in,
                              void* d_out, int out_size, void* d_ws, size_t ws_size,
                              hipStream_t stream) {
    (void)in_sizes; (void)n_in; (void)out_size; (void)ws_size;
    const int*   feats = (const int*)d_in[0];
    const int*   src   = (const int*)d_in[1];
    const int*   dst   = (const int*)d_in[2];
    const float* emb   = (const float*)d_in[3];
    const float* W1    = (const float*)d_in[4];
    const float* b1    = (const float*)d_in[5];
    const float* W2    = (const float*)d_in[6];
    const float* b2    = (const float*)d_in[7];
    float* outp = (float*)d_out;

    // workspace carve-up
    const size_t NF = (size_t)N_NODES * IN_FEATS;       // 6.4M floats
    float* h    = (float*)d_ws;
    float* agg  = h + NF;
    float* so   = agg + NF;          // rsqrt(out_deg)
    float* si   = so + N_NODES;      // rsqrt(in_deg)
    _Float16* W1p = (_Float16*)(si + N_NODES);          // packed: 8 ntiles * 2048 halves
    _Float16* W2p = W1p + 8 * 2048;                     // packed: 3 ntiles * 2048 halves

    const int B = 256;

    // degrees
    zero_f32_kernel<<<(2 * N_NODES + B - 1) / B, B, 0, stream>>>(so, 2 * N_NODES);
    degree_kernel<<<(N_EDGES + B - 1) / B, B, 0, stream>>>(src, dst, so, si, N_EDGES);
    finalize_deg_kernel<<<(N_NODES + B - 1) / B, B, 0, stream>>>(so, si, N_NODES);

    // pack weights into WMMA B-fragment layout (f16, W2 padded 40 -> 48 cols)
    pack_w_kernel<<<(8 * 2048 + B - 1) / B, B, 0, stream>>>(W1, W1p, IN_FEATS, 8);
    pack_w_kernel<<<(3 * 2048 + B - 1) / B, B, 0, stream>>>(W2, W2p, N_CLASSES, 3);

    // embedding mean-pool -> h
    embed_pool_kernel<<<N_NODES, IN_FEATS, 0, stream>>>(feats, emb, h);

    // ---- layer 1: scatter + GEMM(relu) ----
    zero_f32_kernel<<<(int)((NF + B - 1) / B), B, 0, stream>>>(agg, (int)NF);
    scatter_kernel<<<(N_EDGES + 7) / 8, B, 0, stream>>>(src, dst, h, so, agg, N_EDGES);
    // Nw = 128 -> 8 N-tiles -> 256-thread blocks (8 waves), 3125 M-tile blocks
    gemm_wmma_kernel<<<N_NODES / 16, 32 * 8, 0, stream>>>(
        agg, si, (const v16h*)W1p, b1, h, IN_FEATS, IN_FEATS, /*relu=*/1);

    // ---- layer 2: scatter + GEMM ----
    zero_f32_kernel<<<(int)((NF + B - 1) / B), B, 0, stream>>>(agg, (int)NF);
    scatter_kernel<<<(N_EDGES + 7) / 8, B, 0, stream>>>(src, dst, h, so, agg, N_EDGES);
    // Nw = 48 -> 3 N-tiles -> 96-thread blocks
    gemm_wmma_kernel<<<N_NODES / 16, 32 * 3, 0, stream>>>(
        agg, si, (const v16h*)W2p, b2, outp, N_CLASSES, N_CLASSES, /*relu=*/0);
}
